// GATGraph_69011534512671
// MI455X (gfx1250) — matrix operand order repaired and checked by reading, not compiled, not run on hardware
//
#include <hip/hip_runtime.h>
#include <hip/hip_bf16.h>

// ---------------------------------------------------------------------------
// GAT graph network forward for MI455X (gfx1250, wave32).
// - All intermediates in d_ws (L2-resident: 192MB L2 swallows the whole
//   working set, so edge passes are L2-bandwidth/atomic bound, not HBM bound).
// - All N x 64 @ 64 x 64 GEMMs (feat@W per GAT layer, fc_mu first layer) run
//   on the matrix pipe via v_wmma_f32_16x16x32_bf16, fragments fed from LDS
//   with vectorized ds_load_b128 (W staged transposed for contiguous K runs).
// - Edge phase: 3 streaming passes with f32 atomics (atomic-throughput bound).
// ---------------------------------------------------------------------------

typedef __attribute__((ext_vector_type(16))) __bf16 v16bf;
typedef __attribute__((ext_vector_type(8)))  __bf16 v8bf;
typedef __attribute__((ext_vector_type(8)))  float  v8f;

__device__ __forceinline__ float siluf(float x) { return x / (1.f + __expf(-x)); }

__device__ __forceinline__ void atomicMaxF(float* addr, float val) {
  int* ai = (int*)addr;
  int old = *ai;
  while (__int_as_float(old) < val) {
    int assumed = old;
    old = atomicCAS(ai, assumed, __float_as_int(val));
    if (old == assumed) break;
  }
}

// ---------------------------------------------------------------------------
// Node embedding MLPs: feat = concat(mlp_x(x), mlp_h(h))  [N,64]
// ---------------------------------------------------------------------------
__global__ void node_feat_kernel(const float* __restrict__ x, const float* __restrict__ h,
                                 const float* __restrict__ Wx1, const float* __restrict__ bx1,
                                 const float* __restrict__ Wx2, const float* __restrict__ bx2,
                                 const float* __restrict__ Wh1, const float* __restrict__ bh1,
                                 const float* __restrict__ Wh2, const float* __restrict__ bh2,
                                 float* __restrict__ feat, int n) {
  __shared__ float sWx1[3 * 64], sbx1[64], sWx2[64 * 32], sbx2[32];
  __shared__ float sWh1[5 * 64], sbh1[64], sWh2[64 * 32], sbh2[32];
  int tid = threadIdx.x;
  for (int i = tid; i < 3 * 64; i += blockDim.x) sWx1[i] = Wx1[i];
  for (int i = tid; i < 5 * 64; i += blockDim.x) sWh1[i] = Wh1[i];
  for (int i = tid; i < 64 * 32; i += blockDim.x) { sWx2[i] = Wx2[i]; sWh2[i] = Wh2[i]; }
  for (int i = tid; i < 64; i += blockDim.x) { sbx1[i] = bx1[i]; sbh1[i] = bh1[i]; }
  for (int i = tid; i < 32; i += blockDim.x) { sbx2[i] = bx2[i]; sbh2[i] = bh2[i]; }
  __syncthreads();
  int nidx = blockIdx.x * blockDim.x + tid;
  if (nidx >= n) return;

  float xin[3];
  #pragma unroll
  for (int k = 0; k < 3; ++k) xin[k] = x[nidx * 3 + k];
  float hid[64];
  for (int j = 0; j < 64; ++j) {
    float s = sbx1[j];
    #pragma unroll
    for (int k = 0; k < 3; ++k) s += xin[k] * sWx1[k * 64 + j];
    hid[j] = siluf(s);
  }
  for (int j = 0; j < 32; ++j) {
    float s = sbx2[j];
    for (int k = 0; k < 64; ++k) s += hid[k] * sWx2[k * 32 + j];
    feat[(size_t)nidx * 64 + j] = s;
  }
  float hin[5];
  #pragma unroll
  for (int k = 0; k < 5; ++k) hin[k] = h[nidx * 5 + k];
  for (int j = 0; j < 64; ++j) {
    float s = sbh1[j];
    #pragma unroll
    for (int k = 0; k < 5; ++k) s += hin[k] * sWh1[k * 64 + j];
    hid[j] = siluf(s);
  }
  for (int j = 0; j < 32; ++j) {
    float s = sbh2[j];
    for (int k = 0; k < 64; ++k) s += hid[k] * sWh2[k * 32 + j];
    feat[(size_t)nidx * 64 + 32 + j] = s;
  }
}

// ---------------------------------------------------------------------------
// Degree + self-loop edge_attr (fill_value='mean')
// ---------------------------------------------------------------------------
__global__ void zero2_kernel(float* a, float* b, int n) {
  int i = blockIdx.x * blockDim.x + threadIdx.x;
  if (i < n) { a[i] = 0.f; b[i] = 0.f; }
}

__global__ void deg_kernel(const int* __restrict__ dst, const float* __restrict__ ea,
                           float* deg, float* asum, int E) {
  int e = blockIdx.x * blockDim.x + threadIdx.x;
  if (e >= E) return;
  int d = dst[e];
  atomicAdd(&deg[d], 1.f);
  atomicAdd(&asum[d], ea[e]);
}

__global__ void loopattr_kernel(const float* __restrict__ deg, float* la, int n) {
  int i = blockIdx.x * blockDim.x + threadIdx.x;
  if (i < n) la[i] = la[i] / fmaxf(deg[i], 1.f);
}

// ---------------------------------------------------------------------------
// out[n,64] = A[n,64] @ W[64,64]  via v_wmma_f32_16x16x32_bf16.
// Block = 128 threads = 4 waves; block covers 64 rows, each wave owns a
// 16-row band and sweeps all 4 column tiles (8 WMMAs/wave, A frags reused).
// W is staged TRANSPOSED in LDS so every lane's 16 B values are two
// contiguous 16B runs -> ds_load_b128 pairs instead of 16x ds_load_u16.
// Fragment packing per ISA 7.12.2 (wave32):
//   A (16-bit 16x32): lane<16 -> K base 0, lane>=16 -> K base 8; halves 0..7
//     cover K base..base+7, halves 8..15 cover K base+16..base+23.
//   B (32x16): same K packing as A (lane<16 -> K base 0, lane>=16 -> K base 8), column = lane&15.
//   C/D (f32 16x16): VGPR r -> M = r + (lane>=16 ? 8 : 0), N = lane&15.
// ---------------------------------------------------------------------------
__global__ void wmma_gemm64_kernel(const float* __restrict__ A, const float* __restrict__ W,
                                   float* __restrict__ out, int n) {
  __shared__ __bf16 sA[64 * 64];    // row-major rows of A for this block
  __shared__ __bf16 sBt[64 * 64];   // sBt[col*64 + k] = W[k*64 + col]
  int tid = threadIdx.x;
  int row0 = blockIdx.x * 64;
  for (int i = tid; i < 64 * 64; i += 128) {
    sBt[i] = (__bf16)W[((i & 63) << 6) + (i >> 6)];
    int g = row0 + (i >> 6);
    sA[i] = (__bf16)((g < n) ? A[(size_t)g * 64 + (i & 63)] : 0.f);
  }
  __syncthreads();

  int wave = tid >> 5;              // 16-row band
  int lane = tid & 31;
  int m = lane & 15;                // row within band / column within tile
  int aK0 = (lane < 16) ? 0 : 8;
  int bK0 = (lane < 16) ? 0 : 8;

  // A fragments for K blocks 0..31 and 32..63 (loaded once, reused 4x)
  v16bf afrag[2];
  #pragma unroll
  for (int kb = 0; kb < 2; ++kb) {
    int base = (wave * 16 + m) * 64 + kb * 32 + aK0;   // 16B aligned
    v8bf lo = *(const v8bf*)&sA[base];
    v8bf hi = *(const v8bf*)&sA[base + 16];
    afrag[kb] = __builtin_shufflevector(lo, hi,
        0, 1, 2, 3, 4, 5, 6, 7, 8, 9, 10, 11, 12, 13, 14, 15);
  }

  int mo = (lane >= 16) ? 8 : 0;
  int rbase = row0 + wave * 16 + mo;
  bool full = (row0 + 64) <= n;

  #pragma unroll
  for (int ct = 0; ct < 4; ++ct) {
    v8f acc = {};
    #pragma unroll
    for (int kb = 0; kb < 2; ++kb) {
      int base = (ct * 16 + m) * 64 + kb * 32 + bK0;   // 16B aligned
      v8bf lo = *(const v8bf*)&sBt[base];
      v8bf hi = *(const v8bf*)&sBt[base + 16];
      v16bf b = __builtin_shufflevector(lo, hi,
          0, 1, 2, 3, 4, 5, 6, 7, 8, 9, 10, 11, 12, 13, 14, 15);
      acc = __builtin_amdgcn_wmma_f32_16x16x32_bf16(false, afrag[kb], false, b,
                                                    (short)0, acc, false, false);
    }
    if (full) {
      #pragma unroll
      for (int r = 0; r < 8; ++r)
        out[(size_t)(rbase + r) * 64 + ct * 16 + m] = acc[r];
    } else {
      #pragma unroll
      for (int r = 0; r < 8; ++r)
        if (rbase + r < n) out[(size_t)(rbase + r) * 64 + ct * 16 + m] = acc[r];
    }
  }
}

// ---------------------------------------------------------------------------
// a_s/a_d per node + reset softmax accumulators
// ---------------------------------------------------------------------------
__global__ void asad_init_kernel(const float* __restrict__ xw,
                                 const float* __restrict__ asrc, const float* __restrict__ adst,
                                 float* a_s, float* a_d, float* maxb, float* sumb, int n) {
  __shared__ float ss[64], sd[64];
  int tid = threadIdx.x;
  for (int i = tid; i < 64; i += blockDim.x) { ss[i] = asrc[i]; sd[i] = adst[i]; }
  __syncthreads();
  int nidx = blockIdx.x * blockDim.x + tid;
  if (nidx >= n) return;
  float s = 0.f, d = 0.f;
  for (int k = 0; k < 64; ++k) {
    float v = xw[(size_t)nidx * 64 + k];
    s += v * ss[k];
    d += v * sd[k];
  }
  a_s[nidx] = s;
  a_d[nidx] = d;
  maxb[nidx] = -3.0e38f;
  sumb[nidx] = 0.f;
}

__global__ void bias_init_kernel(float* nf, const float* __restrict__ b, int n) {
  int i = blockIdx.x * blockDim.x + threadIdx.x;
  if (i < n * 64) nf[i] = b[i & 63];
}

// c = dot(We[layer], aedge[layer])
__global__ void aedge_dot_kernel(const float* __restrict__ We, const float* __restrict__ aedge,
                                 float* out) {
  __shared__ float red[64];
  int t = threadIdx.x;
  red[t] = We[t] * aedge[t];
  __syncthreads();
  for (int s = 32; s > 0; s >>= 1) {
    if (t < s) red[t] += red[t + s];
    __syncthreads();
  }
  if (t == 0) *out = red[0];
}

// Pass 1: alpha = leaky_relu(a_s[src] + a_d[dst] + a_e, 0.2); segment max
__global__ void alpha_kernel(const int* __restrict__ src, const int* __restrict__ dst,
                             const float* __restrict__ ea, const float* __restrict__ lattr,
                             const float* __restrict__ a_s, const float* __restrict__ a_d,
                             const float* __restrict__ cptr,
                             float* abuf, float* maxb, int E, int n) {
  int e = blockIdx.x * blockDim.x + threadIdx.x;
  if (e >= E + n) return;
  int s, d; float av;
  if (e < E) { s = src[e]; d = dst[e]; av = ea[e]; }
  else       { s = e - E;  d = s;      av = lattr[s]; }
  float al = a_s[s] + a_d[d] + av * (*cptr);
  al = (al > 0.f) ? al : 0.2f * al;
  abuf[e] = al;
  atomicMaxF(&maxb[d], al);
}

// Pass 2: e = exp(alpha - max[dst]); segment sum
__global__ void expsum_kernel(const int* __restrict__ dst, float* abuf,
                              const float* __restrict__ maxb, float* sumb, int E, int n) {
  int e = blockIdx.x * blockDim.x + threadIdx.x;
  if (e >= E + n) return;
  int d = (e < E) ? dst[e] : (e - E);
  float w = __expf(abuf[e] - maxb[d]);
  abuf[e] = w;
  atomicAdd(&sumb[d], w);
}

// Pass 3: feat_out[dst] += xw[src] * (e / sum[dst]).  16 lanes per edge,
// float4 gather from L2-resident xw, 4 f32 atomics per lane.
__global__ void scatter_kernel(const int* __restrict__ src, const int* __restrict__ dst,
                               const float* __restrict__ abuf, const float* __restrict__ sumb,
                               const float* __restrict__ xw, float* nf, int E, int n) {
  long t = (long)blockIdx.x * blockDim.x + threadIdx.x;
  if (t >= (long)(E + n) * 16) return;
  int e = (int)(t >> 4);
  int q = (int)(t & 15);
  int s, d;
  if (e < E) { s = src[e]; d = dst[e]; }
  else       { s = e - E;  d = s; }
  float coef = abuf[e] / fmaxf(sumb[d], 1e-16f);
  const float4 v = *(const float4*)(xw + (size_t)s * 64 + q * 4);
  float* o = nf + (size_t)d * 64 + q * 4;
  __builtin_prefetch(o, 1, 0);   // global_prefetch_b8 toward the atomic target
  atomicAdd(o + 0, v.x * coef);
  atomicAdd(o + 1, v.y * coef);
  atomicAdd(o + 2, v.z * coef);
  atomicAdd(o + 3, v.w * coef);
}

// ---------------------------------------------------------------------------
// fc_mu tail: hid = silu(g + bm1); out = hid @ Wm2 + bm2  (g from WMMA GEMM)
// ---------------------------------------------------------------------------
__global__ void fc_mu_tail_kernel(const float* __restrict__ g,
                                  const float* __restrict__ bm1,
                                  const float* __restrict__ Wm2, const float* __restrict__ bm2,
                                  float* __restrict__ out, int n) {
  __shared__ float sW2[64 * 24], sb2[24], sb1[64];
  int tid = threadIdx.x;
  for (int i = tid; i < 64 * 24; i += blockDim.x) sW2[i] = Wm2[i];
  for (int i = tid; i < 64; i += blockDim.x) sb1[i] = bm1[i];
  for (int i = tid; i < 24; i += blockDim.x) sb2[i] = bm2[i];
  __syncthreads();
  int nidx = blockIdx.x * blockDim.x + tid;
  if (nidx >= n) return;
  float hid[64];
  for (int j = 0; j < 64; ++j) hid[j] = siluf(g[(size_t)nidx * 64 + j] + sb1[j]);
  for (int o = 0; o < 24; ++o) {
    float s = sb2[o];
    for (int j = 0; j < 64; ++j) s += hid[j] * sW2[j * 24 + o];
    out[(size_t)nidx * 24 + o] = s;
  }
}

// ---------------------------------------------------------------------------
extern "C" void kernel_launch(void* const* d_in, const int* in_sizes, int n_in,
                              void* d_out, int out_size, void* d_ws, size_t ws_size,
                              hipStream_t stream) {
  const float* x         = (const float*)d_in[0];
  const float* h         = (const float*)d_in[1];
  const float* edge_attr = (const float*)d_in[2];
  const int*   edge_idx  = (const int*)d_in[3];
  const float* Wx1 = (const float*)d_in[4];
  const float* bx1 = (const float*)d_in[5];
  const float* Wx2 = (const float*)d_in[6];
  const float* bx2 = (const float*)d_in[7];
  const float* Wh1 = (const float*)d_in[8];
  const float* bh1 = (const float*)d_in[9];
  const float* Wh2 = (const float*)d_in[10];
  const float* bh2 = (const float*)d_in[11];
  const float* gat_W     = (const float*)d_in[12];
  const float* gat_asrc  = (const float*)d_in[13];
  const float* gat_adst  = (const float*)d_in[14];
  const float* gat_aedge = (const float*)d_in[15];
  const float* gat_We    = (const float*)d_in[16];
  const float* gat_b     = (const float*)d_in[17];
  const float* Wm1 = (const float*)d_in[18];
  const float* bm1 = (const float*)d_in[19];
  const float* Wm2 = (const float*)d_in[20];
  const float* bm2 = (const float*)d_in[21];

  const int N = in_sizes[0] / 3;
  const int E = in_sizes[2];
  const int* src = edge_idx;
  const int* dst = edge_idx + E;

  // Workspace layout (all L2-resident)
  float* ws = (float*)d_ws;
  size_t NB = (size_t)N * 64;
  float* featA = ws;
  float* featB = featA + NB;
  float* xw    = featB + NB;
  float* a_s   = xw + NB;
  float* a_d   = a_s + N;
  float* deg   = a_d + N;
  float* lattr = deg + N;        // edge-attr sum -> loop_attr (in place)
  float* maxb  = lattr + N;
  float* sumb  = maxb + N;
  float* abuf  = sumb + N;       // E+N attention scratch
  float* cdot  = abuf + (size_t)(E + N);

  auto cdiv = [](long a, long b) { return (int)((a + b - 1) / b); };

  node_feat_kernel<<<cdiv(N, 128), 128, 0, stream>>>(
      x, h, Wx1, bx1, Wx2, bx2, Wh1, bh1, Wh2, bh2, featA, N);
  zero2_kernel<<<cdiv(N, 256), 256, 0, stream>>>(deg, lattr, N);
  deg_kernel<<<cdiv(E, 256), 256, 0, stream>>>(dst, edge_attr, deg, lattr, E);
  loopattr_kernel<<<cdiv(N, 256), 256, 0, stream>>>(deg, lattr, N);

  float* fin = featA;
  float* fout = featB;
  for (int l = 0; l < 3; ++l) {
    const float* W     = gat_W + (size_t)l * 64 * 64;
    const float* asrc  = gat_asrc + l * 64;
    const float* adst  = gat_adst + l * 64;
    const float* aedge = gat_aedge + l * 64;
    const float* We    = gat_We + l * 64;
    const float* gb    = gat_b + l * 64;

    wmma_gemm64_kernel<<<cdiv(N, 64), 128, 0, stream>>>(fin, W, xw, N);
    asad_init_kernel<<<cdiv(N, 128), 128, 0, stream>>>(xw, asrc, adst, a_s, a_d, maxb, sumb, N);
    bias_init_kernel<<<cdiv((long)N * 64, 256), 256, 0, stream>>>(fout, gb, N);
    aedge_dot_kernel<<<1, 64, 0, stream>>>(We, aedge, cdot);
    alpha_kernel<<<cdiv(E + N, 256), 256, 0, stream>>>(
        src, dst, edge_attr, lattr, a_s, a_d, cdot, abuf, maxb, E, N);
    expsum_kernel<<<cdiv(E + N, 256), 256, 0, stream>>>(dst, abuf, maxb, sumb, E, N);
    scatter_kernel<<<cdiv((long)(E + N) * 16, 256), 256, 0, stream>>>(
        src, dst, abuf, sumb, xw, fout, E, N);

    float* t = fin; fin = fout; fout = t;
  }

  // fc_mu: first 64x64 layer on the matrix pipe (xw buffer is free now)
  wmma_gemm64_kernel<<<cdiv(N, 64), 128, 0, stream>>>(fin, Wm1, xw, N);
  fc_mu_tail_kernel<<<cdiv(N, 128), 128, 0, stream>>>(xw, bm1, Wm2, bm2, (float*)d_out, N);
}
